// LSTMBaseline_83545703842183
// MI455X (gfx1250) — matrix operand order repaired and checked by reading, not compiled, not run on hardware
//
#include <hip/hip_runtime.h>
#include <hip/hip_bf16.h>

typedef __attribute__((ext_vector_type(16))) _Float16 v16h;
typedef __attribute__((ext_vector_type(8)))  _Float16 v8h;
typedef __attribute__((ext_vector_type(8)))  float    v8f;
typedef __attribute__((ext_vector_type(4)))  float    v4f;

#define BATCH 64
#define TD 20
#define TQ 20
#define TA 10
#define L784 784
#define DMODEL 256
#define LN_EPS 1e-5f

__device__ __forceinline__ float sigm(float x) { return 1.f / (1.f + __expf(-x)); }

__device__ __forceinline__ float dot4(v4f a, v4f b) {
    return a[0] * b[0] + a[1] * b[1] + a[2] * b[2] + a[3] * b[3];
}

// ---------------------------------------------------------------------------
// Kernel 1: conv(3x3, 1->128) + ReLU, then (784x128)@(128x128 kv_w^T)+bias
// via v_wmma_f32_16x16x32_f16, then LayerNorm -> kvln[t][b][l][e] (f16).
// One workgroup per (t,b); 8 waves; each wave owns a 16-col N-tile.
// ---------------------------------------------------------------------------
__global__ __launch_bounds__(256) void conv_kv_kernel(
    const float* __restrict__ x, const float* __restrict__ conv_w,
    const float* __restrict__ conv_b, const float* __restrict__ kv_w,
    const float* __restrict__ kv_b, const float* __restrict__ ln_g,
    const float* __restrict__ ln_b, _Float16* __restrict__ kvln)
{
    __shared__ float img[30 * 30];          // zero-padded 28x28
    __shared__ float cw[128 * 9];
    __shared__ float cb[128];
    __shared__ _Float16 Ash[112 * 128];     // f16 A row-tile (112 pixels x 128 ch)
    __shared__ float stats[112][2];         // per-row {sum, sumsq} -> {mu, rstd}

    const int tb = blockIdx.x;              // t*BATCH + b
    const int t = tb / BATCH, b = tb % BATCH;
    const int tid = threadIdx.x;
    const int lane = tid & 31, w = tid >> 5;
    const int g = lane >> 4, lm = lane & 15;

    for (int i = tid; i < 128 * 9; i += 256) cw[i] = conv_w[i];
    if (tid < 128) cb[tid] = conv_b[tid];

    const float* xim = x + ((size_t)b * TD + t) * L784;
    for (int i = tid; i < 900; i += 256) {
        int yy = i / 30, xx = i % 30;
        float v = 0.f;
        if (yy >= 1 && yy <= 28 && xx >= 1 && xx <= 28)
            v = xim[(yy - 1) * 28 + (xx - 1)];
        img[i] = v;
    }

    const int N = 16 * w + lm;              // per-lane output column
    const float gl = ln_g[N], bl = ln_b[N], kb = kv_b[N];

    // B fragments: B[k][n] = kv_w[n][k]; elem e, lane l -> K = 32kk+16*(l>=16)+e
    v16h bf[4];
    for (int kk = 0; kk < 4; ++kk) {
        const float* p = kv_w + (size_t)N * 128 + 32 * kk + 16 * g;
        for (int e = 0; e < 16; ++e) bf[kk][e] = (_Float16)p[e];
    }
    __syncthreads();

    _Float16* kout = kvln + (size_t)tb * L784 * 128;

    for (int tile = 0; tile < 7; ++tile) {
        if (tid < 224) ((float*)stats)[tid] = 0.f;
        for (int i = tid; i < 112 * 128; i += 256) {
            int r = i >> 7, ch = i & 127;
            int pix = tile * 112 + r;
            int py = pix / 28 + 1, px = pix % 28 + 1;
            const float* wp = cw + ch * 9;
            float a = cb[ch];
            a += wp[0] * img[(py - 1) * 30 + px - 1] + wp[1] * img[(py - 1) * 30 + px] + wp[2] * img[(py - 1) * 30 + px + 1];
            a += wp[3] * img[py * 30 + px - 1]       + wp[4] * img[py * 30 + px]       + wp[5] * img[py * 30 + px + 1];
            a += wp[6] * img[(py + 1) * 30 + px - 1] + wp[7] * img[(py + 1) * 30 + px] + wp[8] * img[(py + 1) * 30 + px + 1];
            Ash[i] = (_Float16)(a > 0.f ? a : 0.f);
        }
        __syncthreads();

        v8f acc[7];
        for (int m = 0; m < 7; ++m)
            for (int j = 0; j < 8; ++j) acc[m][j] = 0.f;
        for (int kk = 0; kk < 4; ++kk) {
            for (int m = 0; m < 7; ++m) {
                // A 16x32 f16: M = lane&15, K = 32kk + 16*(l>=16) + e
                const int row = 16 * m + lm;
                v8h c0 = *(const v8h*)&Ash[row * 128 + 32 * kk + 16 * g];
                v8h c1 = *(const v8h*)&Ash[row * 128 + 32 * kk + 16 * g + 8];
                v16h af;
                for (int e = 0; e < 8; ++e) { af[e] = c0[e]; af[e + 8] = c1[e]; }
                acc[m] = __builtin_amdgcn_wmma_f32_16x16x32_f16(
                    false, af, false, bf[kk], (short)0, acc[m], false, false);
            }
        }

        // + kv_b, LayerNorm stats; C layout: VGPR j -> M = j + 8*(lane>=16)
        for (int m = 0; m < 7; ++m) {
            for (int j = 0; j < 8; ++j) {
                float v = acc[m][j] + kb;
                acc[m][j] = v;
                float s = v, sq = v * v;
                for (int d = 1; d < 16; d <<= 1) {
                    s  += __shfl_xor(s, d, 32);
                    sq += __shfl_xor(sq, d, 32);
                }
                if (lm == 0) {
                    int row = 16 * m + j + 8 * g;
                    atomicAdd(&stats[row][0], s);
                    atomicAdd(&stats[row][1], sq);
                }
            }
        }
        __syncthreads();
        if (tid < 112) {
            float mu  = stats[tid][0] * (1.f / 128.f);
            float var = stats[tid][1] * (1.f / 128.f) - mu * mu;
            stats[tid][0] = mu;
            stats[tid][1] = rsqrtf(var + LN_EPS);
        }
        __syncthreads();
        for (int m = 0; m < 7; ++m) {
            for (int j = 0; j < 8; ++j) {
                int row = 16 * m + j + 8 * g;
                float v = (acc[m][j] - stats[row][0]) * stats[row][1] * gl + bl;
                kout[(size_t)(tile * 112 + row) * 128 + N] = (_Float16)v;
            }
        }
        __syncthreads();
    }
}

// ---------------------------------------------------------------------------
// Kernel 2: question-phase kv: LN(emb @ kv_w^T + kv_b)
// ---------------------------------------------------------------------------
__global__ __launch_bounds__(128) void kvq_kernel(
    const int* __restrict__ z, const float* __restrict__ index_emb,
    const float* __restrict__ op_emb, const float* __restrict__ kv_w,
    const float* __restrict__ kv_b, const float* __restrict__ ln_g,
    const float* __restrict__ ln_b, float* __restrict__ kvq)
{
    __shared__ float emb[128];
    __shared__ float ss[2];
    const int blk = blockIdx.x, t = blk / BATCH, b = blk % BATCH;
    const int tid = threadIdx.x;
    const int zv = z[b * TQ + t];
    const float* er = (((t % 10) < 5) ? index_emb : op_emb) + (size_t)zv * 128;
    emb[tid] = er[tid];
    if (tid < 2) ss[tid] = 0.f;
    __syncthreads();
    const v4f* wr = (const v4f*)(kv_w + (size_t)tid * 128);
    const v4f* ev = (const v4f*)emb;
    float y = kv_b[tid];
    for (int k = 0; k < 32; ++k) y += dot4(ev[k], wr[k]);
    atomicAdd(&ss[0], y);
    atomicAdd(&ss[1], y * y);
    __syncthreads();
    float mu = ss[0] * (1.f / 128.f);
    float rs = rsqrtf(ss[1] * (1.f / 128.f) - mu * mu + LN_EPS);
    kvq[(size_t)blk * 128 + tid] = (y - mu) * rs * ln_g[tid] + ln_b[tid];
}

// ---------------------------------------------------------------------------
// Kernel 3 (fused recurrence): the whole 50-step loop is independent per
// batch element -> one persistent workgroup (1024 thr = 32 waves) per b.
// h/c state lives in LDS; kvln[t][b] staged to LDS once per digit step;
// attention scores computed with v_wmma (A = padded 16x128 query proj);
// ctx computed with 8-wide LDS vector loads + deterministic partial reduce.
// ---------------------------------------------------------------------------
#define SM_KV    0                       // 784*128 f16  = 200704 B
#define SM_SC    200704                  // 8*784  f32   =  25088 B
#define SM_PART  (SM_SC + 25088)         // 8*8*128 f32  =  32768 B
#define SM_QP16  (SM_PART + 32768)       // 16*128 f16   =   4096 B
#define SM_CTX   (SM_QP16 + 4096)        // 8*128  f32   =   4096 B
#define SM_G     (SM_CTX + 4096)         // 1024   f32   =   4096 B
#define SM_H     (SM_G + 4096)           // 256    f32
#define SM_C     (SM_H + 1024)           // 256    f32
#define SM_X     (SM_C + 1024)           // 128    f32
#define SM_Q     (SM_X + 512)            // 128    f32
#define SM_Q2    (SM_Q + 512)            // 128    f32
#define SM_O     (SM_Q2 + 512)           // 128    f32
#define SM_QB    (SM_O + 512)            // 8      f32
#define SM_LG    (SM_QB + 32)            // 16     f32
#define SM_TOTAL (SM_LG + 64)            // ~275 KB < 320 KB/WGP

__global__ __launch_bounds__(1024) void recurrence_kernel(
    const _Float16* __restrict__ kvln, const float* __restrict__ kvq,
    const float* __restrict__ q_w, const float* __restrict__ q_b,
    const float* __restrict__ in_proj_w, const float* __restrict__ in_proj_b,
    const float* __restrict__ attn_out_w, const float* __restrict__ attn_out_b,
    const float* __restrict__ w_ih, const float* __restrict__ w_hh,
    const float* __restrict__ b_ih, const float* __restrict__ b_hh,
    const float* __restrict__ h0, const float* __restrict__ c0,
    const float* __restrict__ out_w, const float* __restrict__ out_b,
    float* __restrict__ out)
{
    extern __shared__ char smem[];
    _Float16* kvsh = (_Float16*)(smem + SM_KV);
    float*    sc   = (float*)   (smem + SM_SC);
    float*    part = (float*)   (smem + SM_PART);
    _Float16* qp16 = (_Float16*)(smem + SM_QP16);
    float*    ctx  = (float*)   (smem + SM_CTX);
    float*    gsh  = (float*)   (smem + SM_G);
    float*    hsh  = (float*)   (smem + SM_H);
    float*    csh  = (float*)   (smem + SM_C);
    float*    xsh  = (float*)   (smem + SM_X);
    float*    qsh  = (float*)   (smem + SM_Q);
    float*    q2   = (float*)   (smem + SM_Q2);
    float*    osh  = (float*)   (smem + SM_O);
    float*    qb2  = (float*)   (smem + SM_QB);
    float*    lg   = (float*)   (smem + SM_LG);

    const int b = blockIdx.x, tid = threadIdx.x;
    const int lane = tid & 31, wid = tid >> 5;
    const int g = lane >> 4, lm = lane & 15;

    if (tid < 256) { hsh[tid] = h0[tid]; csh[tid] = c0[tid]; }
    __syncthreads();

    for (int step = 0; step < TD + TQ + TA; ++step) {
        if (step < TD) {
            // ---- stage kvln[t][b] (200KB f16) into LDS, coalesced 16B ----
            const _Float16* kvb = kvln + ((size_t)step * BATCH + b) * L784 * 128;
            for (int i = tid; i < (L784 * 128) / 8; i += 1024)
                *(v8h*)(kvsh + i * 8) = *(const v8h*)(kvb + i * 8);

            // ---- query path: q = (h@q_w^T+q_b)@wq^T+bq; fold wk + scale ----
            if (tid < 128) {
                float a = q_b[tid];
                const v4f* wr = (const v4f*)(q_w + (size_t)tid * 256);
                const v4f* hv = (const v4f*)hsh;
                for (int k = 0; k < 64; ++k) a += dot4(hv[k], wr[k]);
                qsh[tid] = a;
            }
            __syncthreads();
            if (tid < 128) {
                float a = in_proj_b[tid];                 // bq; wq = rows [0,128)
                const v4f* wr = (const v4f*)(in_proj_w + (size_t)tid * 128);
                const v4f* qv = (const v4f*)qsh;
                for (int k = 0; k < 32; ++k) a += dot4(qv[k], wr[k]);
                q2[tid] = a;
            }
            __syncthreads();
            {   // qp16[h][e] = 0.25 * sum_d q2[h*16+d]*wk[h*16+d][e]; rows 8..15 = 0
                const int hh = tid >> 7, e = tid & 127;
                float acc = 0.f;
                for (int d = 0; d < 16; ++d)
                    acc += q2[hh * 16 + d] * in_proj_w[(size_t)(128 + hh * 16 + d) * 128 + e];
                qp16[hh * 128 + e] = (_Float16)(acc * 0.25f);
                qp16[(hh + 8) * 128 + e] = (_Float16)0.f;
            }
            if (tid < 8) {
                float acc = 0.f;
                for (int d = 0; d < 16; ++d)
                    acc += q2[tid * 16 + d] * in_proj_b[128 + tid * 16 + d];
                qb2[tid] = acc * 0.25f;
            }
            __syncthreads();

            // ---- scores via WMMA: D[h][l] = sum_e qp16[h][e]*kvsh[l][e] ----
            v16h af[4];
            for (int kk = 0; kk < 4; ++kk) {
                v8h c0v = *(const v8h*)&qp16[lm * 128 + 32 * kk + 16 * g];
                v8h c1v = *(const v8h*)&qp16[lm * 128 + 32 * kk + 16 * g + 8];
                for (int e = 0; e < 8; ++e) { af[kk][e] = c0v[e]; af[kk][e + 8] = c1v[e]; }
            }
            for (int tile = wid; tile < 49; tile += 32) {
                const int l0 = tile * 16;
                v8f acc;
                for (int j = 0; j < 8; ++j) acc[j] = 0.f;
                for (int kk = 0; kk < 4; ++kk) {
                    // B[k][n]=kvsh[l=n][e=k]: lane -> l = l0+lm, elems K = 32kk+16g+e
                    v8h b0 = *(const v8h*)&kvsh[(l0 + lm) * 128 + 32 * kk + 16 * g];
                    v8h b1 = *(const v8h*)&kvsh[(l0 + lm) * 128 + 32 * kk + 16 * g + 8];
                    v16h bfr;
                    for (int e = 0; e < 8; ++e) { bfr[e] = b0[e]; bfr[e + 8] = b1[e]; }
                    acc = __builtin_amdgcn_wmma_f32_16x16x32_f16(
                        false, af[kk], false, bfr, (short)0, acc, false, false);
                }
                if (g == 0)   // C VGPR j -> head j (heads 8..15 are padding)
                    for (int j = 0; j < 8; ++j) sc[j * L784 + l0 + lm] = acc[j] + qb2[j];
            }
            __syncthreads();

            // ---- softmax over l: one wave per head ----
            if (wid < 8) {
                float* row = sc + wid * L784;
                float m = -3.4e38f;
                for (int l = lane; l < L784; l += 32) m = fmaxf(m, row[l]);
                for (int d = 1; d < 32; d <<= 1) m = fmaxf(m, __shfl_xor(m, d, 32));
                float s = 0.f;
                for (int l = lane; l < L784; l += 32) {
                    float e_ = __expf(row[l] - m);
                    row[l] = e_;
                    s += e_;
                }
                for (int d = 1; d < 32; d <<= 1) s += __shfl_xor(s, d, 32);
                float inv = 1.f / s;
                for (int l = lane; l < L784; l += 32) row[l] *= inv;
            }
            __syncthreads();

            // ---- ctx[h][e] = sum_l a[h][l]*kvsh[l][e] ----
            // thread = (head hh, l-partition lp of 8, e-block eb of 8 halfs)
            {
                const int hh = tid >> 7, sub = tid & 127;
                const int lp = sub >> 4, eb = sub & 15;
                const float* arow = sc + hh * L784;
                float acc[8];
                for (int j = 0; j < 8; ++j) acc[j] = 0.f;
                const int l0 = lp * 98, l1 = l0 + 98;     // 8*98 = 784
                for (int l = l0; l < l1; ++l) {
                    v8h kvv = *(const v8h*)&kvsh[l * 128 + eb * 8];
                    float a = arow[l];
                    for (int j = 0; j < 8; ++j) acc[j] += a * (float)kvv[j];
                }
                float* pp = part + (size_t)(lp * 8 + hh) * 128 + eb * 8;
                for (int j = 0; j < 8; ++j) pp[j] = acc[j];
            }
            __syncthreads();
            {   // deterministic partial reduction over the 8 l-partitions
                const int hh = tid >> 7, e = tid & 127;
                float s = 0.f;
                for (int p = 0; p < 8; ++p) s += part[(size_t)(p * 8 + hh) * 128 + e];
                ctx[hh * 128 + e] = s;
            }
            __syncthreads();

            // o[e'] = ctx[h(e')].wv[e'] + bv[e']   (wv = in_proj_w rows [256,384))
            if (tid < 128) {
                const int hh = tid >> 4;
                float acc = in_proj_b[256 + tid];
                const v4f* wr = (const v4f*)(in_proj_w + (size_t)(256 + tid) * 128);
                const v4f* cv = (const v4f*)(ctx + hh * 128);
                for (int k = 0; k < 32; ++k) acc += dot4(cv[k], wr[k]);
                osh[tid] = acc;
            }
            __syncthreads();
            if (tid < 128) {  // lstm_in = o @ attn_out_w^T + attn_out_b
                float acc = attn_out_b[tid];
                const v4f* wr = (const v4f*)(attn_out_w + (size_t)tid * 128);
                const v4f* ov = (const v4f*)osh;
                for (int k = 0; k < 32; ++k) acc += dot4(ov[k], wr[k]);
                xsh[tid] = acc;
            }
            __syncthreads();
        } else {
            if (tid < 128) {
                float v = 0.f;
                if (step < TD + TQ)
                    v = kvq[((size_t)(step - TD) * BATCH + b) * 128 + tid];
                xsh[tid] = v;
            }
            __syncthreads();
        }

        // ---- LSTM: 1024 threads, one gate row each, float4 streams ----
        {
            const int r = tid;
            float a = b_ih[r] + b_hh[r];
            const v4f* wi4 = (const v4f*)(w_ih + (size_t)r * 128);
            const v4f* wh4 = (const v4f*)(w_hh + (size_t)r * 256);
            const v4f* xv = (const v4f*)xsh;
            const v4f* hv = (const v4f*)hsh;
            for (int k = 0; k < 32; ++k) a += dot4(xv[k], wi4[k]);
            for (int k = 0; k < 64; ++k) a += dot4(hv[k], wh4[k]);
            gsh[r] = a;
        }
        __syncthreads();
        if (tid < 256) {
            const int d = tid;
            float cn = sigm(gsh[256 + d]) * csh[d] + sigm(gsh[d]) * tanhf(gsh[512 + d]);
            float hv = sigm(gsh[768 + d]) * tanhf(cn);
            csh[d] = cn;
            hsh[d] = hv;
        }
        __syncthreads();

        // ---- predict: logits + entropy certainty ----
        if (tid < 10) {
            float acc = out_b[tid];
            const v4f* wr = (const v4f*)(out_w + (size_t)tid * 256);
            const v4f* hv = (const v4f*)hsh;
            for (int k = 0; k < 64; ++k) acc += dot4(hv[k], wr[k]);
            lg[tid] = acc;
            out[b * 500 + tid * 50 + step] = acc;        // predictions (B,10,50)
        }
        __syncthreads();
        if (tid == 0) {
            float m = lg[0];
            for (int i = 1; i < 10; ++i) m = fmaxf(m, lg[i]);
            float s = 0.f;
            for (int i = 0; i < 10; ++i) s += __expf(lg[i] - m);
            float lse = m + __logf(s);
            float ne = 0.f;
            for (int i = 0; i < 10; ++i) { float lp = lg[i] - lse; ne -= __expf(lp) * lp; }
            ne *= (1.f / 2.302585092994046f);            // / ln(10)
            float* cert = out + BATCH * 10 * 50;         // certainties (B,2,50)
            cert[b * 100 + step] = ne;
            cert[b * 100 + 50 + step] = 1.f - ne;
        }
        __syncthreads();
    }
}

// ---------------------------------------------------------------------------
extern "C" void kernel_launch(void* const* d_in, const int* in_sizes, int n_in,
                              void* d_out, int out_size, void* d_ws, size_t ws_size,
                              hipStream_t stream)
{
    (void)in_sizes; (void)n_in; (void)out_size; (void)ws_size;
    const float* x          = (const float*)d_in[0];
    const int*   z          = (const int*)  d_in[1];
    const float* conv_w     = (const float*)d_in[2];
    const float* conv_b     = (const float*)d_in[3];
    const float* index_emb  = (const float*)d_in[4];
    const float* op_emb     = (const float*)d_in[5];
    const float* kv_w       = (const float*)d_in[6];
    const float* kv_b       = (const float*)d_in[7];
    const float* ln_g       = (const float*)d_in[8];
    const float* ln_b       = (const float*)d_in[9];
    const float* q_w        = (const float*)d_in[10];
    const float* q_b        = (const float*)d_in[11];
    const float* in_proj_w  = (const float*)d_in[12];
    const float* in_proj_b  = (const float*)d_in[13];
    const float* attn_out_w = (const float*)d_in[14];
    const float* attn_out_b = (const float*)d_in[15];
    const float* w_ih       = (const float*)d_in[16];
    const float* w_hh       = (const float*)d_in[17];
    const float* b_ih       = (const float*)d_in[18];
    const float* b_hh       = (const float*)d_in[19];
    const float* h0         = (const float*)d_in[20];
    const float* c0         = (const float*)d_in[21];
    const float* out_w      = (const float*)d_in[22];
    const float* out_b      = (const float*)d_in[23];
    float* out = (float*)d_out;

    char* ws = (char*)d_ws;
    size_t off = 0;
    _Float16* kvln = (_Float16*)(ws + off);
    off += (size_t)TD * BATCH * L784 * 128 * sizeof(_Float16);   // 256.9 MB
    float* kvq = (float*)(ws + off);
    off += (size_t)TQ * BATCH * 128 * sizeof(float);

    // Phase A: all digit-phase kvln (WMMA GEMM + LN) and question-phase kv
    conv_kv_kernel<<<TD * BATCH, 256, 0, stream>>>(x, conv_w, conv_b, kv_w, kv_b,
                                                   ln_g, ln_b, kvln);
    kvq_kernel<<<TQ * BATCH, 128, 0, stream>>>(z, index_emb, op_emb, kv_w, kv_b,
                                               ln_g, ln_b, kvq);
    // Phase B: fully fused 50-step recurrence, one persistent WG per batch elem
    recurrence_kernel<<<BATCH, 1024, SM_TOTAL, stream>>>(
        kvln, kvq, q_w, q_b, in_proj_w, in_proj_b, attn_out_w, attn_out_b,
        w_ih, w_hh, b_ih, b_hh, h0, c0, out_w, out_b, out);
}
